// AttnDecoderRNN_79070347919689
// MI455X (gfx1250) — matrix operand order repaired and checked by reading, not compiled, not run on hardware
//
#include <hip/hip_runtime.h>
#include <hip/hip_bf16.h>
#include <stdint.h>

// Problem dimensions (match reference)
#define H_   1024
#define E_   300
#define EP_  320            // E padded to multiple of 32 for WMMA K
#define V_   32
#define S_   128
#define B_   256
#define T_   10
#define KCAT (EP_ + H_)     // 1344 = concat K for [x | h] @ [w_ih | w_hh]^T
#define G4H  (4 * H_)       // 4096

#define LDS_STRIDE 40       // 32 bf16 + 8 pad (80 B rows -> spread LDS banks)

typedef __bf16 bf16_t;
typedef bf16_t v16bf __attribute__((ext_vector_type(16)));
typedef float  v8f   __attribute__((ext_vector_type(8)));
typedef unsigned short u16;

union BFrag { v16bf v; uint4 q[2]; };

__device__ __forceinline__ u16 f2bf(float x) {
    union { float f; unsigned int u; } c; c.f = x;
    unsigned int r = (c.u >> 16) & 1u;
    c.u += 0x7fffu + r;                 // round-to-nearest-even
    return (u16)(c.u >> 16);
}
__device__ __forceinline__ float bf2f(u16 h) {
    union { unsigned int u; float f; } c; c.u = ((unsigned int)h) << 16;
    return c.f;
}
__device__ __forceinline__ float sigmoidf_(float x) { return 1.0f / (1.0f + __expf(-x)); }

__device__ __forceinline__ v8f wmma_bf16(v16bf a, v16bf b, v8f c) {
    return __builtin_amdgcn_wmma_f32_16x16x32_bf16(false, a, false, b, (short)0, c, false, false);
}

// A fragment (16x32 bf16) from a row-major buffer: lane m,sel -> K [sel*8..+7] and [16+sel*8..+7]
__device__ __forceinline__ v16bf frag_a(const u16* p0, int stride, int lane) {
    const int m = lane & 15, sel = lane >> 4;
    const u16* p = p0 + (size_t)m * stride;
    BFrag f;
    f.q[0] = *reinterpret_cast<const uint4*>(p + sel * 8);
    f.q[1] = *reinterpret_cast<const uint4*>(p + 16 + sel * 8);
    return f.v;
}
// B fragment (32x16 bf16) from row-major W (row n = B column n): lane n,sel -> K [sel*8..+7] and [16+sel*8..+7]
__device__ __forceinline__ v16bf frag_b(const u16* p0, int stride, int lane) {
    const int n = lane & 15, sel = lane >> 4;
    const u16* p = p0 + (size_t)n * stride + sel * 8;
    BFrag f;
    f.q[0] = *reinterpret_cast<const uint4*>(p);
    f.q[1] = *reinterpret_cast<const uint4*>(p + 16);
    return f.v;
}

__device__ __forceinline__ void store_tile(v8f acc, float* Cf, u16* Cbf, int ldc,
                                           int rowbase, int colbase, const float* bias, int lane) {
    const int n = lane & 15, sel = lane >> 4;
    const int col = colbase + n;
    const float bv = bias ? bias[col] : 0.0f;
#pragma unroll
    for (int r = 0; r < 8; ++r) {
        const int row = rowbase + r + sel * 8;
        const float v = acc[r] + bv;
        if (Cf)  Cf [(size_t)row * ldc + col] = v;
        if (Cbf) Cbf[(size_t)row * ldc + col] = f2bf(v);
    }
}

// Async-copy a 64x32 bf16 tile (row-major, leading dim ld) into LDS (stride LDS_STRIDE).
// 256 x 16B chunks, 128 threads x 2. Lands directly in LDS, tracked by ASYNCcnt.
__device__ __forceinline__ void async_copy_tile64x32(const u16* gbase, int ld,
                                                     unsigned lds_base, int tid) {
#pragma unroll
    for (int it = 0; it < 2; ++it) {
        const int c   = tid + it * 128;
        const int row = c >> 2, cc = c & 3;
        const u16* g  = gbase + (size_t)row * ld + cc * 8;
        const unsigned l = lds_base + (unsigned)(row * LDS_STRIDE + cc * 8) * 2;
        asm volatile("global_load_async_to_lds_b128 %0, %1, off"
                     :: "v"(l), "v"(g) : "memory");
    }
}
__device__ __forceinline__ void wait_async0() {
    asm volatile("s_wait_asynccnt 0x0" ::: "memory");
}

// ---------------------------------------------------------------------------
// LDS-tiled bf16 WMMA GEMM: C[M,N] = A[M,K] @ W[N,K]^T + bias[N]
// 128 threads = 4 waves in 2x2, block tile 64x64, wave tile 32x32 (2x2 frags).
// Double-buffered LDS stages filled by global_load_async_to_lds_b128.
// Requires M%64==0, N%64==0, K%32==0.
// ---------------------------------------------------------------------------
__global__ __launch_bounds__(128)
void wmma_gemm_bf16_lds(const u16* __restrict__ A, int lda,
                        const u16* __restrict__ W, int ldb,
                        const float* __restrict__ bias,
                        float* __restrict__ Cf, u16* __restrict__ Cbf,
                        int ldc, int K)
{
    __shared__ u16 sA[2 * 64 * LDS_STRIDE];
    __shared__ u16 sW[2 * 64 * LDS_STRIDE];

    const int tid  = threadIdx.x;
    const int lane = tid & 31;
    const int wv   = tid >> 5;
    const int wm   = wv >> 1, wn = wv & 1;
    const int ntile = blockIdx.x * 64;
    const int mtile = blockIdx.y * 64;

    const unsigned sAb = (unsigned)(size_t)(void*)sA;
    const unsigned sWb = (unsigned)(size_t)(void*)sW;
    const unsigned bufBytes = 64 * LDS_STRIDE * 2;

    const u16* Ab = A + (size_t)mtile * lda;
    const u16* Wb = W + (size_t)ntile * ldb;

    v8f acc[2][2] = {};

    const int nk = K / 32;
    // prologue: fill buffer 0
    async_copy_tile64x32(Ab, lda, sAb, tid);
    async_copy_tile64x32(Wb, ldb, sWb, tid);

    for (int kt = 0; kt < nk; ++kt) {
        wait_async0();          // this wave's async fills complete
        __syncthreads();        // all waves' fills complete & prior reads done
        const int buf = kt & 1;
        if (kt + 1 < nk) {      // prefetch next K-slab into the other buffer
            const int k2 = (kt + 1) * 32;
            async_copy_tile64x32(Ab + k2, lda, sAb + (buf ^ 1) * bufBytes, tid);
            async_copy_tile64x32(Wb + k2, ldb, sWb + (buf ^ 1) * bufBytes, tid);
        }
        const u16* a0 = sA + buf * 64 * LDS_STRIDE + (wm * 32) * LDS_STRIDE;
        const u16* w0 = sW + buf * 64 * LDS_STRIDE + (wn * 32) * LDS_STRIDE;
        const v16bf af0 = frag_a(a0, LDS_STRIDE, lane);
        const v16bf af1 = frag_a(a0 + 16 * LDS_STRIDE, LDS_STRIDE, lane);
        const v16bf bf0 = frag_b(w0, LDS_STRIDE, lane);
        const v16bf bf1 = frag_b(w0 + 16 * LDS_STRIDE, LDS_STRIDE, lane);
        acc[0][0] = wmma_bf16(af0, bf0, acc[0][0]);
        acc[0][1] = wmma_bf16(af0, bf1, acc[0][1]);
        acc[1][0] = wmma_bf16(af1, bf0, acc[1][0]);
        acc[1][1] = wmma_bf16(af1, bf1, acc[1][1]);
        // WMMA's implicit dscnt wait means this wave's LDS reads are complete
        // before it signals the next iteration's barrier.
    }

#pragma unroll
    for (int ti = 0; ti < 2; ++ti)
#pragma unroll
        for (int tj = 0; tj < 2; ++tj)
            store_tile(acc[ti][tj], Cf, Cbf, ldc,
                       mtile + wm * 32 + ti * 16,
                       ntile + wn * 32 + tj * 16, bias, lane);
}

// ---------------------------------------------------------------------------
// Direct (no-LDS) bf16 WMMA GEMM for small N (one wave per 16x16 tile)
// ---------------------------------------------------------------------------
__global__ __launch_bounds__(32)
void wmma_gemm_bf16(const u16* __restrict__ A, int lda,
                    const u16* __restrict__ W, int ldb,
                    const float* __restrict__ bias,
                    float* __restrict__ Cf, u16* __restrict__ Cbf,
                    int ldc, int N, int K)
{
    const int lane  = threadIdx.x & 31;
    const int m16   = lane & 15;
    const int sel   = lane >> 4;
    const int ntile = blockIdx.x * 16;
    const int mtile = blockIdx.y * 16;
    if (ntile >= N) return;

    const u16* arow = A + (size_t)(mtile + m16) * lda;
    const u16* wrow = W + (size_t)(ntile + m16) * ldb;

    v8f acc = {};
    for (int k = 0; k < K; k += 32) {
        BFrag a, b;
        a.q[0] = *reinterpret_cast<const uint4*>(arow + k + sel * 8);
        a.q[1] = *reinterpret_cast<const uint4*>(arow + k + 16 + sel * 8);
        b.q[0] = *reinterpret_cast<const uint4*>(wrow + k + sel * 8);
        b.q[1] = *reinterpret_cast<const uint4*>(wrow + k + 16 + sel * 8);
        acc = wmma_bf16(a.v, b.v, acc);
    }
    store_tile(acc, Cf, Cbf, ldc, mtile, ntile, bias, lane);
}

// ---------------------------------------------------------------------------
// Prep kernels
// ---------------------------------------------------------------------------
__global__ void k_f32_to_bf16(const float* __restrict__ src,
                              u16* __restrict__ dst, size_t n)
{
    size_t i = (size_t)blockIdx.x * blockDim.x + threadIdx.x;
    size_t stride = (size_t)gridDim.x * blockDim.x;
    for (; i < n; i += stride) dst[i] = f2bf(src[i]);
}

__global__ void k_build_wcat(const float* __restrict__ w_ih,
                             const float* __restrict__ w_hh,
                             u16* __restrict__ wcat)
{
    size_t total = (size_t)G4H * KCAT;
    size_t i = (size_t)blockIdx.x * blockDim.x + threadIdx.x;
    size_t stride = (size_t)gridDim.x * blockDim.x;
    for (; i < total; i += stride) {
        int n = (int)(i / KCAT), k = (int)(i % KCAT);
        float v;
        if (k < E_)        v = w_ih[(size_t)n * E_ + k];
        else if (k < EP_)  v = 0.0f;
        else               v = w_hh[(size_t)n * H_ + (k - EP_)];
        wcat[i] = f2bf(v);
    }
}

__global__ void k_build_emb_pad(const float* __restrict__ emb,
                                u16* __restrict__ emb_pad)
{
    int i = blockIdx.x * blockDim.x + threadIdx.x;
    if (i >= V_ * EP_) return;
    int v = i / EP_, k = i % EP_;
    emb_pad[i] = (k < E_) ? f2bf(emb[(size_t)v * E_ + k]) : (u16)0;
}

__global__ void k_bias_sum(const float* __restrict__ b_ih,
                           const float* __restrict__ b_hh,
                           float* __restrict__ bias)
{
    int i = blockIdx.x * blockDim.x + threadIdx.x;
    if (i < G4H) bias[i] = b_ih[i] + b_hh[i];
}

__global__ void k_init(const float* __restrict__ h0, const float* __restrict__ c0,
                       float* __restrict__ h, float* __restrict__ c,
                       u16* __restrict__ acat,
                       float* __restrict__ state, int* __restrict__ tok)
{
    int i = blockIdx.x * blockDim.x + threadIdx.x;
    if (i < B_ * H_) {
        float hv = h0[i];
        h[i] = hv; c[i] = c0[i];
        int b = i / H_, k = i % H_;
        acat[(size_t)b * KCAT + EP_ + k] = f2bf(hv);
    }
    if (i < B_ * 3) state[i] = 0.0f;
    if (i < B_)     tok[i] = 0;
}

// ---------------------------------------------------------------------------
// Per-step kernels
// ---------------------------------------------------------------------------
__global__ void k_gather_x(const int* __restrict__ tok,
                           const u16* __restrict__ emb_pad,
                           u16* __restrict__ acat)
{
    int i = blockIdx.x * blockDim.x + threadIdx.x;
    if (i >= B_ * EP_) return;
    int b = i / EP_, k = i % EP_;
    acat[(size_t)b * KCAT + k] = emb_pad[(size_t)tok[b] * EP_ + k];
}

__global__ void k_lstm_act(const float* __restrict__ gates,
                           float* __restrict__ h, float* __restrict__ c,
                           u16* __restrict__ acat,
                           u16* __restrict__ catCH)
{
    int i = blockIdx.x * blockDim.x + threadIdx.x;
    if (i >= B_ * H_) return;
    int b = i / H_, k = i % H_;
    const float* g = gates + (size_t)b * G4H;
    float ig = sigmoidf_(g[k]);
    float fg = sigmoidf_(g[H_ + k]);
    float gg = tanhf(g[2 * H_ + k]);
    float og = sigmoidf_(g[3 * H_ + k]);
    float cn = fg * c[i] + ig * gg;
    float hn = og * tanhf(cn);
    c[i] = cn; h[i] = hn;
    u16 hb = f2bf(hn);
    acat [(size_t)b * KCAT + EP_ + k]    = hb;  // next-step LSTM input half
    catCH[(size_t)b * (2 * H_) + H_ + k] = hb;  // [context | h] for output head
}

__global__ __launch_bounds__(256)
void k_attn_scores(const u16* __restrict__ enc_t_bf,
                   const float* __restrict__ dec_t,
                   const float* __restrict__ attn_w, const float* __restrict__ attn_b,
                   const int* __restrict__ lens, float* __restrict__ scores)
{
    int wid  = (int)((blockIdx.x * blockDim.x + threadIdx.x) >> 5);
    int lane = threadIdx.x & 31;
    if (wid >= S_ * B_) return;
    int s = wid & (S_ - 1);
    int b = wid >> 7;           // S_ == 128
    const u16* et = enc_t_bf + ((size_t)s * B_ + b) * H_;
    const float* dt = dec_t + (size_t)b * H_;
    float acc = 0.0f;
    for (int k = lane; k < H_; k += 32)
        acc += tanhf(bf2f(et[k]) + dt[k]) * attn_w[k];
#pragma unroll
    for (int m = 16; m > 0; m >>= 1) acc += __shfl_xor(acc, m, 32);
    if (lane == 0)
        scores[(size_t)b * S_ + s] = (s < lens[b]) ? (acc + attn_b[0]) : -3.4028235e38f;
}

__global__ __launch_bounds__(256)
void k_softmax_context(const float* __restrict__ scores,
                       const u16* __restrict__ enc_bf,
                       float* __restrict__ ctx_out,          // [B,H] (already offset to t)
                       u16* __restrict__ catCH)
{
    __shared__ float red[256];
    __shared__ float p[S_];
    int b = blockIdx.x, tid = threadIdx.x;
    float v = (tid < S_) ? scores[(size_t)b * S_ + tid] : -3.4028235e38f;
    red[tid] = v; __syncthreads();
    for (int s = 128; s > 0; s >>= 1) {
        if (tid < s) red[tid] = fmaxf(red[tid], red[tid + s]);
        __syncthreads();
    }
    float mx = red[0]; __syncthreads();
    float e = (tid < S_) ? __expf(v - mx) : 0.0f;
    red[tid] = e; __syncthreads();
    for (int s = 128; s > 0; s >>= 1) {
        if (tid < s) red[tid] += red[tid + s];
        __syncthreads();
    }
    float inv = 1.0f / red[0];
    if (tid < S_) p[tid] = e * inv;
    __syncthreads();

    float a0 = 0.f, a1 = 0.f, a2 = 0.f, a3 = 0.f;
    const u16* base = enc_bf + (size_t)b * H_;
    for (int s = 0; s < S_; ++s) {
        float w = p[s];
        const u16* ep = base + (size_t)s * B_ * H_;
        a0 += w * bf2f(ep[tid]);
        a1 += w * bf2f(ep[tid + 256]);
        a2 += w * bf2f(ep[tid + 512]);
        a3 += w * bf2f(ep[tid + 768]);
    }
    ctx_out[(size_t)b * H_ + tid      ] = a0;
    ctx_out[(size_t)b * H_ + tid + 256] = a1;
    ctx_out[(size_t)b * H_ + tid + 512] = a2;
    ctx_out[(size_t)b * H_ + tid + 768] = a3;
    catCH[(size_t)b * (2 * H_) + tid      ] = f2bf(a0);
    catCH[(size_t)b * (2 * H_) + tid + 256] = f2bf(a1);
    catCH[(size_t)b * (2 * H_) + tid + 512] = f2bf(a2);
    catCH[(size_t)b * (2 * H_) + tid + 768] = f2bf(a3);
}

__global__ __launch_bounds__(256)
void k_decode(const float* __restrict__ logits,
              const float* __restrict__ asm_w, const float* __restrict__ asm_b,
              const float* __restrict__ asm_p,
              float* __restrict__ state, int* __restrict__ tok,
              float* __restrict__ probs_out)   // [B,V] (already offset to t)
{
    int lane = threadIdx.x & 31;
    int b = blockIdx.x * 8 + (threadIdx.x >> 5);
    float lg = logits[(size_t)b * V_ + lane];
    float mx = lg;
#pragma unroll
    for (int m = 16; m > 0; m >>= 1) mx = fmaxf(mx, __shfl_xor(mx, m, 32));
    float e = __expf(lg - mx);
    float sum = e;
#pragma unroll
    for (int m = 16; m > 0; m >>= 1) sum += __shfl_xor(sum, m, 32);
    float prob = e / sum;
    probs_out[(size_t)b * V_ + lane] = prob;

    float s0 = state[b * 3 + 0], s1 = state[b * 3 + 1], s2 = state[b * 3 + 2];
    float mn = 3.4028235e38f;
#pragma unroll
    for (int f = 0; f < 4; ++f) {
        float tv = s0 * asm_w[(0 * V_ + lane) * 4 + f]
                 + s1 * asm_w[(1 * V_ + lane) * 4 + f]
                 + s2 * asm_w[(2 * V_ + lane) * 4 + f]
                 - asm_b[lane * 4 + f];
        mn = fminf(mn, tv);
    }
    float vp = (mn > 0.0f ? 1.0f : 0.0f) * prob;
    float mx2 = vp;
#pragma unroll
    for (int m = 16; m > 0; m >>= 1) mx2 = fmaxf(mx2, __shfl_xor(mx2, m, 32));
    float e2 = __expf(vp - mx2);
    float sum2 = e2;
#pragma unroll
    for (int m = 16; m > 0; m >>= 1) sum2 += __shfl_xor(sum2, m, 32);
    float p2 = e2 / sum2;

    // argmax, first-max wins (matches jnp.argmax)
    int bi = lane; float bv = p2;
#pragma unroll
    for (int m = 16; m > 0; m >>= 1) {
        float ov = __shfl_xor(bv, m, 32);
        int   oi = __shfl_xor(bi, m, 32);
        if (ov > bv || (ov == bv && oi < bi)) { bv = ov; bi = oi; }
    }
    if (lane == 0) tok[b] = bi;
    if (lane < 3) state[b * 3 + lane] += asm_p[bi * 3 + lane];
}

__global__ void k_finalize(const float* __restrict__ h, const float* __restrict__ c,
                           const float* __restrict__ state,
                           float* __restrict__ out_h, float* __restrict__ out_c,
                           float* __restrict__ out_state)
{
    int i = blockIdx.x * blockDim.x + threadIdx.x;
    if (i < B_ * H_) { out_h[i] = h[i]; out_c[i] = c[i]; }
    if (i < B_ * 3) out_state[i] = state[i];
}

// ---------------------------------------------------------------------------
// Host orchestration
// ---------------------------------------------------------------------------
extern "C" void kernel_launch(void* const* d_in, const int* in_sizes, int n_in,
                              void* d_out, int out_size, void* d_ws, size_t ws_size,
                              hipStream_t stream)
{
    const float* enc    = (const float*)d_in[0];
    const int*   lens   = (const int*)  d_in[1];
    const float* h0     = (const float*)d_in[2];
    const float* c0     = (const float*)d_in[3];
    const float* emb    = (const float*)d_in[4];
    const float* w_ih   = (const float*)d_in[5];
    const float* w_hh   = (const float*)d_in[6];
    const float* b_ih   = (const float*)d_in[7];
    const float* b_hh   = (const float*)d_in[8];
    const float* enc_W  = (const float*)d_in[9];
    const float* enc_b  = (const float*)d_in[10];
    const float* dec_W  = (const float*)d_in[11];
    const float* dec_b  = (const float*)d_in[12];
    const float* attn_w = (const float*)d_in[13];
    const float* attn_b = (const float*)d_in[14];
    const float* out_W  = (const float*)d_in[15];
    const float* out_b  = (const float*)d_in[16];
    const float* asm_w  = (const float*)d_in[17];
    const float* asm_b  = (const float*)d_in[18];
    const float* asm_p  = (const float*)d_in[19];

    // workspace layout
    char* ws = (char*)d_ws;
    size_t off = 0;
    auto alloc = [&](size_t bytes) { size_t o = off; off += (bytes + 255) & ~(size_t)255; return o; };
    u16*   enc_bf   = (u16*)(ws + alloc((size_t)S_ * B_ * H_ * 2)); // 64 MB
    u16*   enc_t_bf = (u16*)(ws + alloc((size_t)S_ * B_ * H_ * 2)); // 64 MB
    u16*   wcat     = (u16*)(ws + alloc((size_t)G4H * KCAT * 2));
    u16*   wdec     = (u16*)(ws + alloc((size_t)H_ * H_ * 2));
    u16*   wenc     = (u16*)(ws + alloc((size_t)H_ * H_ * 2));
    u16*   wout     = (u16*)(ws + alloc((size_t)V_ * 2 * H_ * 2));
    u16*   emb_pad  = (u16*)(ws + alloc((size_t)V_ * EP_ * 2));
    float* bias_sum = (float*)(ws + alloc((size_t)G4H * 4));
    float* h_buf    = (float*)(ws + alloc((size_t)B_ * H_ * 4));
    float* c_buf    = (float*)(ws + alloc((size_t)B_ * H_ * 4));
    u16*   acat     = (u16*)(ws + alloc((size_t)B_ * KCAT * 2));
    float* gates    = (float*)(ws + alloc((size_t)B_ * G4H * 4));
    float* dec_t    = (float*)(ws + alloc((size_t)B_ * H_ * 4));
    float* scores   = (float*)(ws + alloc((size_t)B_ * S_ * 4));
    u16*   catCH    = (u16*)(ws + alloc((size_t)B_ * 2 * H_ * 2));
    float* logits   = (float*)(ws + alloc((size_t)B_ * V_ * 4));
    int*   tok      = (int*)(ws + alloc((size_t)B_ * 4));
    float* state    = (float*)(ws + alloc((size_t)B_ * 3 * 4));

    // output layout: probs_seq [T,B,V], h [B,H], c [B,H], ctx_seq [T,B,H], state [B,3]
    float* out       = (float*)d_out;
    float* out_probs = out;
    float* out_h     = out_probs + (size_t)T_ * B_ * V_;
    float* out_c     = out_h + (size_t)B_ * H_;
    float* out_ctx   = out_c + (size_t)B_ * H_;
    float* out_state = out_ctx + (size_t)T_ * B_ * H_;

    // ---- prep ----
    k_f32_to_bf16<<<4096, 256, 0, stream>>>(enc, enc_bf, (size_t)S_ * B_ * H_);
    k_f32_to_bf16<<<1024, 256, 0, stream>>>(enc_W, wenc, (size_t)H_ * H_);
    k_f32_to_bf16<<<1024, 256, 0, stream>>>(dec_W, wdec, (size_t)H_ * H_);
    k_f32_to_bf16<<<256, 256, 0, stream>>>(out_W, wout, (size_t)V_ * 2 * H_);
    k_build_wcat<<<4096, 256, 0, stream>>>(w_ih, w_hh, wcat);
    k_build_emb_pad<<<(V_ * EP_ + 255) / 256, 256, 0, stream>>>(emb, emb_pad);
    k_bias_sum<<<(G4H + 255) / 256, 256, 0, stream>>>(b_ih, b_hh, bias_sum);
    k_init<<<(B_ * H_ + 255) / 256, 256, 0, stream>>>(h0, c0, h_buf, c_buf, acat, state, tok);

    // enc_t = encoder_outputs @ enc_W^T + enc_b  (M=32768, N=1024, K=1024) -> bf16
    wmma_gemm_bf16_lds<<<dim3(H_ / 64, (S_ * B_) / 64), 128, 0, stream>>>(
        enc_bf, H_, wenc, H_, enc_b, nullptr, enc_t_bf, H_, H_);

    // ---- decode loop ----
    for (int t = 0; t < T_; ++t) {
        k_gather_x<<<(B_ * EP_ + 255) / 256, 256, 0, stream>>>(tok, emb_pad, acat);

        // gates = [x|h] @ [w_ih|w_hh]^T + (b_ih+b_hh)   (M=256, N=4096, K=1344)
        wmma_gemm_bf16_lds<<<dim3(G4H / 64, B_ / 64), 128, 0, stream>>>(
            acat, KCAT, wcat, KCAT, bias_sum, gates, nullptr, G4H, KCAT);

        k_lstm_act<<<(B_ * H_ + 255) / 256, 256, 0, stream>>>(gates, h_buf, c_buf, acat, catCH);

        // dec_t = h @ dec_W^T + dec_b   (M=256, N=1024, K=1024); A = h half of acat
        wmma_gemm_bf16_lds<<<dim3(H_ / 64, B_ / 64), 128, 0, stream>>>(
            acat + EP_, KCAT, wdec, H_, dec_b, dec_t, nullptr, H_, H_);

        k_attn_scores<<<(S_ * B_) / 8, 256, 0, stream>>>(
            enc_t_bf, dec_t, attn_w, attn_b, lens, scores);

        k_softmax_context<<<B_, 256, 0, stream>>>(
            scores, enc_bf, out_ctx + (size_t)t * B_ * H_, catCH);

        // logits = [context|h] @ out_W^T + out_b   (M=256, N=32, K=2048)
        wmma_gemm_bf16<<<dim3(V_ / 16, B_ / 16), 32, 0, stream>>>(
            catCH, 2 * H_, wout, 2 * H_, out_b, logits, nullptr, V_, V_, 2 * H_);

        k_decode<<<B_ / 8, 256, 0, stream>>>(
            logits, asm_w, asm_b, asm_p, state, tok, out_probs + (size_t)t * B_ * V_);
    }

    k_finalize<<<(B_ * H_ + 255) / 256, 256, 0, stream>>>(
        h_buf, c_buf, state, out_h, out_c, out_state);
}
